// ContactGMP_36988258353212
// MI455X (gfx1250) — matrix operand order repaired and checked by reading, not compiled, not run on hardware
//
#include <hip/hip_runtime.h>

#define N_NODES 50000
#define L_FEAT  128
#define E_MAIN  800000
#define E_CONT  200000

typedef __attribute__((ext_vector_type(16))) __bf16 v16bf;
typedef __attribute__((ext_vector_type(8)))  __bf16 v8bf;
typedef __attribute__((ext_vector_type(8)))  float  v8f;

// Wave-private LDS ordering fence (LDS ops are in-order per wave; this stops
// the compiler from reordering and makes the producer->consumer edge explicit).
#define LDS_FENCE()   asm volatile("s_wait_dscnt 0x0" ::: "memory")
// Wait for all outstanding async (memory->LDS) transfers of this wave.
#define ASYNC_FENCE() asm volatile("s_wait_asynccnt 0x0" ::: "memory")

// Per-lane async gather: 16 bytes global -> LDS, tracked with ASYNCcnt.
// dst_lds is the byte offset into the workgroup's LDS space (addrspace(3)
// pointers are 32-bit offsets from LDS base), src is a 64-bit global address.
// VGLOBAL async encoding (ISA 15.18.3 op 98): VDST = LDS addr VGPR,
// VADDR = 64-bit global address pair, SADDR = off (GV mode).
__device__ __forceinline__ void async_gather_b128(unsigned dst_lds, const void* src)
{
  unsigned long long ga = (unsigned long long)src;
  asm volatile("global_load_async_to_lds_b128 %0, %1, off"
               :: "v"(dst_lds), "v"(ga) : "memory");
}

// ---------------------------------------------------------------------------
// One MLP layer: C[16x128] = A[16xK] * Wt^T + b, A in LDS (bf16, row-major,
// stride strideA), Wt in global as [n][K] bf16 (i.e. W transposed), K = 32*nkt.
// Wave32 WMMA fragment layouts used here (A/B K order is this code's contiguous choice):
//   A (16x32 bf16): lane M=lane&15; elem e = K[e + 16*hk], hk = lane>>4.
//   B (32x16 bf16): lane N=lane&15; elem e = K[e + 16*hk].
//   C (16x16 f32):  vgpr k -> row k+8*hk, col lane&15 (+16*nt per N-tile).
// ---------------------------------------------------------------------------
__device__ __forceinline__ void mlp_layer(const __bf16* __restrict__ Abase, int strideA, int nkt,
                                          const __bf16* __restrict__ wt, int sK,
                                          const float* __restrict__ bias,
                                          int lane, v8f acc[8])
{
  const int n0 = lane & 15;
  const int hk = lane >> 4;
#pragma unroll
  for (int nt = 0; nt < 8; ++nt) {
    float bv = bias[n0 + 16 * nt];
#pragma unroll
    for (int k = 0; k < 8; ++k) acc[nt][k] = bv;
  }
  for (int kt = 0; kt < nkt; ++kt) {
    const __bf16* ar = Abase + n0 * strideA + kt * 32 + 16 * hk;
    v8bf alo = *(const v8bf*)(ar);        // K = 16hk .. 16hk+7
    v8bf ahi = *(const v8bf*)(ar + 8);    // K = 16hk+8 .. 16hk+15
    v16bf a;
#pragma unroll
    for (int t = 0; t < 8; ++t) { a[t] = alo[t]; a[t + 8] = ahi[t]; }
#pragma unroll
    for (int nt = 0; nt < 8; ++nt) {
      const __bf16* br = wt + (size_t)(n0 + 16 * nt) * sK + kt * 32 + 16 * hk;
      v16bf b = *(const v16bf*)br;        // 32 bytes, elem e <-> K offset e
      acc[nt] = __builtin_amdgcn_wmma_f32_16x16x32_bf16(
          false, a, false, b, (short)0, acc[nt], false, false);
    }
  }
}

__device__ __forceinline__ void store_relu(__bf16* H, int strideH, int lane, const v8f acc[8])
{
  const int n0 = lane & 15, hk = lane >> 4;
#pragma unroll
  for (int k = 0; k < 8; ++k) {
    const int row = k + 8 * hk;
#pragma unroll
    for (int nt = 0; nt < 8; ++nt) {
      float v = acc[nt][k];
      H[row * strideH + n0 + 16 * nt] = (__bf16)(v > 0.f ? v : 0.f);
    }
  }
}

// Row-wise LayerNorm stats over 128 cols: in-lane sum over 8 N-tiles, then
// butterfly reduce across the 16 lanes that share the row (xor 1,2,4,8 stays
// within each wave half, matching the C-fragment layout).
__device__ __forceinline__ void layernorm_stats(const v8f acc[8], float mean[8], float inv[8])
{
  float rs[8], rq[8];
#pragma unroll
  for (int k = 0; k < 8; ++k) {
    float s = 0.f, q = 0.f;
#pragma unroll
    for (int nt = 0; nt < 8; ++nt) { float v = acc[nt][k]; s += v; q += v * v; }
    rs[k] = s; rq[k] = q;
  }
#pragma unroll
  for (int m = 1; m < 16; m <<= 1) {
#pragma unroll
    for (int k = 0; k < 8; ++k) {
      rs[k] += __shfl_xor(rs[k], m, 32);
      rq[k] += __shfl_xor(rq[k], m, 32);
    }
  }
#pragma unroll
  for (int k = 0; k < 8; ++k) {
    float mu  = rs[k] * (1.f / 128.f);
    float var = rq[k] * (1.f / 128.f) - mu * mu;
    mean[k] = mu;
    inv[k]  = __frsqrt_rn(var + 1e-5f);
  }
}

// ---------------------------------------------------------------------------
// Edge MLP: one wave = 16 edges. Builds [16 x 288] bf16 feature tile in LDS
// (geo feats | x_i | x_j | zero pad) using async global->LDS gathers, runs
// 3 WMMA layers + LN, atomically scatters the 16x128 result into agg[j].
// ---------------------------------------------------------------------------
template<bool USE_GEO>
__global__ __launch_bounds__(128)
void edge_mlp_kernel(const __bf16* __restrict__ xbf,
                     const int* __restrict__ g, int E,
                     const float* __restrict__ pos_posw,
                     const __bf16* __restrict__ wt0, const float* __restrict__ b0,
                     const __bf16* __restrict__ wt1, const float* __restrict__ b1,
                     const __bf16* __restrict__ wt2, const float* __restrict__ b2,
                     float* __restrict__ agg)
{
  __shared__ __align__(32) __bf16 sA[4][16 * 288];
  __shared__ __align__(32) __bf16 sH[4][16 * 128];
  __shared__ int sI[4][16];
  __shared__ int sJ[4][16];

  const int lane = threadIdx.x & 31;
  const int wave = threadIdx.x >> 5;
  const int tile = blockIdx.x * 4 + wave;
  const int e0 = tile * 16;
  if (e0 >= E) return;                 // uniform per wave: EXEC stays all-ones

  __bf16* A = sA[wave];
  __bf16* H = sH[wave];
  int* I = sI[wave];
  int* J = sJ[wave];

  if (lane < 16) {
    const int e = e0 + lane;
    const int i = g[e];
    const int j = g[E + e];
    I[lane] = i; J[lane] = j;
    const float* pi = pos_posw + (size_t)i * 6;
    const float* pj = pos_posw + (size_t)j * 6;
    float f[8];
    int c = 0;
    if (USE_GEO) {
      float dx = pi[0] - pj[0], dy = pi[1] - pj[1], dz = pi[2] - pj[2];
      f[0] = dx; f[1] = dy; f[2] = dz; f[3] = sqrtf(dx*dx + dy*dy + dz*dz);
      c = 4;
    }
    float wx = pi[3] - pj[3], wy = pi[4] - pj[4], wz = pi[5] - pj[5];
    f[c] = wx; f[c+1] = wy; f[c+2] = wz; f[c+3] = sqrtf(wx*wx + wy*wy + wz*wz);
    for (int t = c + 4; t < 8; ++t) f[t] = 0.f;
    __bf16* row = A + lane * 288;
    for (int t = 0; t < 8; ++t)   row[t] = (__bf16)f[t];
    for (int t = 264; t < 288; ++t) row[t] = (__bf16)0.f;   // K pad 264->288
  }
  LDS_FENCE();

  // Gather x_i / x_j rows: 16 rows x (16+16) chunks of 8 bf16 (16B each),
  // pushed straight into LDS with async loads (no VGPR staging).
  for (int u = lane; u < 512; u += 32) {
    const int row  = u >> 5;
    const int part = u & 31;
    const int idx  = (part < 16) ? I[row] : J[row];
    const int c8   = part & 15;
    const void* src = (const void*)(xbf + (size_t)idx * 128 + c8 * 8);
    unsigned dst = (unsigned)(unsigned long long)
        (A + row * 288 + 8 + ((part < 16) ? 0 : 128) + c8 * 8);
    async_gather_b128(dst, src);
  }
  ASYNC_FENCE();

  v8f acc[8];
  mlp_layer(A, 288, 9, wt0, 288, b0, lane, acc);   // K = 288
  store_relu(H, 128, lane, acc);
  LDS_FENCE();
  mlp_layer(H, 128, 4, wt1, 128, b1, lane, acc);   // K = 128
  store_relu(A, 128, lane, acc);                   // reuse A region
  LDS_FENCE();
  mlp_layer(A, 128, 4, wt2, 128, b2, lane, acc);   // K = 128

  float mean[8], inv[8];
  layernorm_stats(acc, mean, inv);

  const int n0 = lane & 15, hk = lane >> 4;
#pragma unroll
  for (int k = 0; k < 8; ++k) {
    const int row = k + 8 * hk;
    const int j   = J[row];
    float* dst = agg + (size_t)j * 128 + n0;
#pragma unroll
    for (int nt = 0; nt < 8; ++nt)
      atomicAdd(dst + 16 * nt, (acc[nt][k] - mean[k]) * inv[k]);
  }
}

// ---------------------------------------------------------------------------
// Node MLP: one wave = 16 nodes, input tile [16 x 384] = [x | agg_m | agg_c].
// Output = LN(mlp(...)) + x. Block = 64 threads (2 waves) to fit LDS.
// ---------------------------------------------------------------------------
__global__ __launch_bounds__(64)
void node_mlp_kernel(const float* __restrict__ x,
                     const __bf16* __restrict__ xbf,
                     const float* __restrict__ aggm,
                     const float* __restrict__ aggc,
                     const __bf16* __restrict__ wt0, const float* __restrict__ b0,
                     const __bf16* __restrict__ wt1, const float* __restrict__ b1,
                     const __bf16* __restrict__ wt2, const float* __restrict__ b2,
                     float* __restrict__ out, int nNodes)
{
  __shared__ __align__(32) __bf16 sA[2][16 * 384];
  __shared__ __align__(32) __bf16 sH[2][16 * 128];

  const int lane = threadIdx.x & 31;
  const int wave = threadIdx.x >> 5;
  const int tile = blockIdx.x * 2 + wave;
  const int node0 = tile * 16;
  if (node0 >= nNodes) return;

  __bf16* A = sA[wave];
  __bf16* H = sH[wave];

  // 16 rows x 48 chunks of 8 elems: [0,16)=xbf copy, [16,32)=agg_m, [32,48)=agg_c
  for (int u = lane; u < 16 * 48; u += 32) {
    const int row  = u / 48;
    const int part = u % 48;
    const int node = node0 + row;
    __bf16* dst = A + row * 384 + part * 8;
    if (part < 16) {
      *(v8bf*)dst = *(const v8bf*)(xbf + (size_t)node * 128 + part * 8);
    } else {
      const float* src = (part < 32) ? (aggm + (size_t)node * 128 + (part - 16) * 8)
                                     : (aggc + (size_t)node * 128 + (part - 32) * 8);
#pragma unroll
      for (int t = 0; t < 8; ++t) dst[t] = (__bf16)src[t];
    }
  }
  LDS_FENCE();

  v8f acc[8];
  mlp_layer(A, 384, 12, wt0, 384, b0, lane, acc);  // K = 384
  store_relu(H, 128, lane, acc);
  LDS_FENCE();
  mlp_layer(H, 128, 4, wt1, 128, b1, lane, acc);
  store_relu(A, 128, lane, acc);
  LDS_FENCE();
  mlp_layer(A, 128, 4, wt2, 128, b2, lane, acc);

  float mean[8], inv[8];
  layernorm_stats(acc, mean, inv);

  const int n0 = lane & 15, hk = lane >> 4;
#pragma unroll
  for (int k = 0; k < 8; ++k) {
    const int row  = k + 8 * hk;
    const int node = node0 + row;
#pragma unroll
    for (int nt = 0; nt < 8; ++nt) {
      const int col = n0 + 16 * nt;
      out[(size_t)node * 128 + col] =
          (acc[nt][k] - mean[k]) * inv[k] + x[(size_t)node * 128 + col];
    }
  }
}

// ---------------------------------------------------------------------------
// Prep kernels: fp32 -> bf16 conversions, weight transpose+pad, zero fill.
// ---------------------------------------------------------------------------
__global__ void cvt_x_kernel(const float* __restrict__ src, __bf16* __restrict__ dst, int n)
{
  int i = blockIdx.x * 256 + threadIdx.x;
  if (i < n) dst[i] = (__bf16)src[i];
}

__global__ void zero_kernel(float* __restrict__ p, int n)
{
  int i = blockIdx.x * 256 + threadIdx.x;
  if (i < n) p[i] = 0.f;
}

// W: [Kin][128] fp32 (reference layout). wt: [128][Kpad] bf16 (B-fragment layout).
// mode 0: identity K mapping (zero pad beyond Kin).
// mode 1: cont-MLP mapping for the shared 288 stride: k<4 -> dw/|dw| rows,
//         4..7 -> zero, 8.. -> x_i/x_j rows shifted by 4.
__global__ void cvt_w_kernel(const float* __restrict__ W, __bf16* __restrict__ wt,
                             int Kin, int Kpad, int mode)
{
  int idx = blockIdx.x * 256 + threadIdx.x;
  if (idx >= 128 * Kpad) return;
  int n = idx / Kpad;
  int k = idx % Kpad;
  int sk;
  if (mode == 0) {
    sk = (k < Kin) ? k : -1;
  } else {
    if (k < 4) sk = k;
    else if (k < 8) sk = -1;
    else { sk = k - 4; if (sk >= Kin) sk = -1; }
  }
  wt[idx] = (sk >= 0) ? (__bf16)W[(size_t)sk * 128 + n] : (__bf16)0.f;
}

// ---------------------------------------------------------------------------
extern "C" void kernel_launch(void* const* d_in, const int* in_sizes, int n_in,
                              void* d_out, int out_size, void* d_ws, size_t ws_size,
                              hipStream_t stream)
{
  (void)in_sizes; (void)n_in; (void)out_size; (void)ws_size;

  const float* x  = (const float*)d_in[0];
  const int*   g  = (const int*)  d_in[1];
  const int*   cg = (const int*)  d_in[2];
  const float* pp = (const float*)d_in[3];
  const float* W[9]; const float* B[9];
  for (int m = 0; m < 3; ++m)
    for (int l = 0; l < 3; ++l) {
      W[m * 3 + l] = (const float*)d_in[4 + m * 6 + l * 2];
      B[m * 3 + l] = (const float*)d_in[4 + m * 6 + l * 2 + 1];
    }
  // W/B[0..2] = main, [3..5] = cont, [6..8] = node

  char* ws = (char*)d_ws;
  size_t off = 0;
  auto take = [&](size_t bytes) -> void* {
    void* p = (void*)(ws + off);
    off += (bytes + 255) & ~(size_t)255;
    return p;
  };

  __bf16* xbf = (__bf16*)take((size_t)N_NODES * 128 * 2);
  float*  aggm = (float*)take((size_t)N_NODES * 128 * 4);
  float*  aggc = (float*)take((size_t)N_NODES * 128 * 4);

  const int Kin[9]  = {264, 128, 128, 260, 128, 128, 384, 128, 128};
  const int Kpad[9] = {288, 128, 128, 288, 128, 128, 384, 128, 128};
  const int mode[9] = {0, 0, 0, 1, 0, 0, 0, 0, 0};
  __bf16* wt[9];
  for (int m = 0; m < 9; ++m) wt[m] = (__bf16*)take((size_t)128 * Kpad[m] * 2);

  const int nx = N_NODES * 128;
  cvt_x_kernel<<<(nx + 255) / 256, 256, 0, stream>>>(x, xbf, nx);
  zero_kernel<<<(nx + 255) / 256, 256, 0, stream>>>(aggm, nx);
  zero_kernel<<<(nx + 255) / 256, 256, 0, stream>>>(aggc, nx);
  for (int m = 0; m < 9; ++m) {
    const int tot = 128 * Kpad[m];
    cvt_w_kernel<<<(tot + 255) / 256, 256, 0, stream>>>(W[m], wt[m], Kin[m], Kpad[m], mode[m]);
  }

  edge_mlp_kernel<true><<<E_MAIN / 16 / 4, 128, 0, stream>>>(
      xbf, g, E_MAIN, pp, wt[0], B[0], wt[1], B[1], wt[2], B[2], aggm);
  edge_mlp_kernel<false><<<E_CONT / 16 / 4, 128, 0, stream>>>(
      xbf, cg, E_CONT, pp, wt[3], B[3], wt[4], B[4], wt[5], B[5], aggc);

  const int ntiles = N_NODES / 16;                 // 3125
  node_mlp_kernel<<<(ntiles + 1) / 2, 64, 0, stream>>>(
      x, xbf, aggm, aggc, wt[6], B[6], wt[7], B[7], wt[8], B[8], (float*)d_out, N_NODES);
}
